// AnyPrecisionLinear_84662395338928
// MI455X (gfx1250) — compile-verified, matrix operand order checked by reading, not run on hardware
//
#include <hip/hip_runtime.h>

typedef __attribute__((ext_vector_type(16))) _Float16 v16h;
typedef __attribute__((ext_vector_type(8)))  float    v8f;

#define IN_F   4096
#define OUT_F  11008
#define WORDS  (IN_F / 32)          // 128 int32 words per row per bitplane
#define PLANE  (OUT_F * WORDS)      // element stride between bitplanes
#define BM     256
#define BN     128
#define BK     64
#define LDA    72                   // halves; 144B rows keep 16B alignment, spread banks
#define LDB    72
#define LDC    136                  // halves; 272B rows, 16B aligned

union Frag {
    v16h v;
    uint4 q[2];
};

// Spread the low 8 bits of t so that bit i lands at bit 4*i.
__device__ __forceinline__ unsigned int spread4(unsigned int t) {
    t = (t | (t << 12)) & 0x000F000Fu;
    t = (t | (t << 6))  & 0x03030303u;
    t = (t | (t << 3))  & 0x11111111u;
    return t;
}

// 4 bitplane words (plane 0 = MSB of code, MSB-first bit order within a word)
// -> 4 dwords of 8 packed 4-bit codes; nibble n of dword d = code for k = 8d+n.
__device__ __forceinline__ void codes32(unsigned int b0, unsigned int b1,
                                        unsigned int b2, unsigned int b3,
                                        unsigned int cz[4]) {
    const unsigned int rb0 = __builtin_bitreverse32(b0);
    const unsigned int rb1 = __builtin_bitreverse32(b1);
    const unsigned int rb2 = __builtin_bitreverse32(b2);
    const unsigned int rb3 = __builtin_bitreverse32(b3);
#pragma unroll
    for (int d = 0; d < 4; ++d) {
        const unsigned int s0 = spread4((rb0 >> (8 * d)) & 0xFFu);
        const unsigned int s1 = spread4((rb1 >> (8 * d)) & 0xFFu);
        const unsigned int s2 = spread4((rb2 >> (8 * d)) & 0xFFu);
        const unsigned int s3 = spread4((rb3 >> (8 * d)) & 0xFFu);
        cz[d] = (s0 << 3) | (s1 << 2) | (s2 << 1) | s3;
    }
}

__device__ __forceinline__ unsigned int lut_pair(const unsigned short* lrow,
                                                 unsigned int codes, int n) {
    const unsigned int c0 = (codes >> (4 * n)) & 0xFu;
    const unsigned int c1 = (codes >> (4 * n + 4)) & 0xFu;
    return (unsigned int)lrow[c0] | ((unsigned int)lrow[c1] << 16);
}

// ---------------------------------------------------------------------------
// Kernel 1: one-shot dequant of the full weight matrix into scratch (fp16,
// row-major [OUT_F][IN_F]). One thread per (row, 32-weight word).
// ---------------------------------------------------------------------------
__global__ __launch_bounds__(256)
void apl_dequant_kernel(const int*      __restrict__ qw,
                        const _Float16* __restrict__ lut,
                        _Float16*       __restrict__ wout)
{
    __shared__ __align__(16) _Float16 lutS[32];   // 2 rows x 16 entries
    const int tid = threadIdx.x;
    const int o   = blockIdx.x * 2 + (tid >> 7);  // output row
    const int wd  = tid & 127;                    // word within row

    if (tid < 32)
        lutS[tid] = lut[(size_t)blockIdx.x * 32 + tid];
    __syncthreads();

    const unsigned short* lrow = (const unsigned short*)lutS + (tid >> 7) * 16;
    const int* q = qw + (size_t)o * WORDS + wd;

    unsigned int cz[4];
    codes32((unsigned int)q[0], (unsigned int)q[PLANE],
            (unsigned int)q[2 * PLANE], (unsigned int)q[3 * PLANE], cz);

    _Float16* dst = wout + (size_t)o * IN_F + wd * 32;
#pragma unroll
    for (int d = 0; d < 4; ++d) {
        uint4 v;
        v.x = lut_pair(lrow, cz[d], 0);
        v.y = lut_pair(lrow, cz[d], 2);
        v.z = lut_pair(lrow, cz[d], 4);
        v.w = lut_pair(lrow, cz[d], 6);
        *(uint4*)(dst + d * 8) = v;
    }
}

// ---------------------------------------------------------------------------
// Kernel 2: pure fp16 WMMA GEMM y = x * W^T + b with W from scratch.
// 256x128 block tile, 8 waves of 64x64, K-tile 64. W stays L2-resident.
// ---------------------------------------------------------------------------
__global__ __launch_bounds__(256)
void apl_gemm_kernel(const _Float16* __restrict__ x,
                     const _Float16* __restrict__ w,
                     const _Float16* __restrict__ bias,
                     _Float16*       __restrict__ out)
{
    __shared__ __align__(16) unsigned char smem[BM * LDA * 2 + BN * LDB * 2];
    _Float16* As = (_Float16*)smem;
    _Float16* Bs = (_Float16*)(smem + BM * LDA * 2);
    _Float16* Cs = (_Float16*)smem;     // 128*136*2 = 34816B <= 36864B

    const int tid  = threadIdx.x;
    const int lane = tid & 31;
    const int wave = tid >> 5;
    const int r    = lane & 15;
    const int hi   = lane >> 4;
    const int wm   = wave & 3;          // 0..3 -> 64 rows each
    const int wn   = wave >> 2;         // 0..1 -> 64 cols each

    const int m0 = blockIdx.y * BM;
    const int n0 = blockIdx.x * BN;

    v8f acc[4][4];
#pragma unroll
    for (int mi = 0; mi < 4; ++mi)
#pragma unroll
        for (int ni = 0; ni < 4; ++ni)
            acc[mi][ni] = (v8f){0.f, 0.f, 0.f, 0.f, 0.f, 0.f, 0.f, 0.f};

#pragma unroll 1
    for (int kt = 0; kt < IN_F; kt += BK) {
        __syncthreads();

        // stage A: 256 rows x 64 halves (b128 coalesced)
#pragma unroll
        for (int it = 0; it < 8; ++it) {
            int lin = it * 256 + tid;
            int row = lin >> 3, ch = lin & 7;
            *(uint4*)(As + row * LDA + ch * 8) =
                *(const uint4*)(x + (size_t)(m0 + row) * IN_F + kt + ch * 8);
        }
        // stage B: 256 rows x 32 halves (b128 coalesced, L2-resident W)
#pragma unroll
        for (int it = 0; it < 4; ++it) {
            int lin = it * 256 + tid;
            int row = lin >> 2, ch = lin & 3;
            *(uint4*)(Bs + row * LDB + ch * 8) =
                *(const uint4*)(w + (size_t)(n0 + row) * IN_F + kt + ch * 8);
        }

        const int ktn = (kt + BK < IN_F) ? (kt + BK) : kt;
        __builtin_prefetch(x + (size_t)(m0 + tid) * IN_F + ktn, 0, 0);
        __builtin_prefetch(w + (size_t)(n0 + (tid & 127)) * IN_F + ktn, 0, 0);

        __syncthreads();

#pragma unroll
        for (int ks = 0; ks < BK; ks += 32) {
            Frag a[4], b[4];
#pragma unroll
            for (int mi = 0; mi < 4; ++mi) {
                const _Float16* ap = As + (wm * 64 + mi * 16 + r) * LDA + ks + hi * 8;
                a[mi].q[0] = *(const uint4*)ap;
                a[mi].q[1] = *(const uint4*)(ap + 16);
            }
#pragma unroll
            for (int ni = 0; ni < 4; ++ni) {
                const _Float16* bp = Bs + (wn * 64 + ni * 16 + r) * LDB + ks + hi * 8;
                b[ni].q[0] = *(const uint4*)bp;
                b[ni].q[1] = *(const uint4*)(bp + 16);
            }
#pragma unroll
            for (int mi = 0; mi < 4; ++mi)
#pragma unroll
                for (int ni = 0; ni < 4; ++ni)
                    acc[mi][ni] = __builtin_amdgcn_wmma_f32_16x16x32_f16(
                        false, a[mi].v, false, b[ni].v,
                        (short)0, acc[mi][ni], false, false);
        }
    }

    // epilogue: bias in f32, two-phase LDS staging, coalesced b128 stores
    float bb[4];
#pragma unroll
    for (int ni = 0; ni < 4; ++ni)
        bb[ni] = (float)bias[n0 + wn * 64 + ni * 16 + r];

#pragma unroll 1
    for (int ph = 0; ph < 2; ++ph) {
        __syncthreads();
        if ((wm >> 1) == ph) {          // wave-uniform
#pragma unroll
            for (int mi = 0; mi < 4; ++mi)
#pragma unroll
                for (int ni = 0; ni < 4; ++ni)
#pragma unroll
                    for (int i = 0; i < 8; ++i) {
                        const int rowl = (wm & 1) * 64 + mi * 16 + hi * 8 + i;
                        const int col  = wn * 64 + ni * 16 + r;
                        Cs[rowl * LDC + col] = (_Float16)(acc[mi][ni][i] + bb[ni]);
                    }
        }
        __syncthreads();
#pragma unroll
        for (int it = 0; it < 8; ++it) {
            int lin = it * 256 + tid;
            int row = lin >> 4, ch = lin & 15;
            *(uint4*)(out + (size_t)(m0 + ph * 128 + row) * OUT_F + n0 + ch * 8) =
                *(const uint4*)(Cs + row * LDC + ch * 8);
        }
    }
}

// ---------------------------------------------------------------------------
// Fallback: fused dequant+GEMM (used when scratch is too small for W).
// ---------------------------------------------------------------------------
__global__ __launch_bounds__(256)
void apl_fused_kernel(const _Float16* __restrict__ x,
                      const int*      __restrict__ qw,
                      const _Float16* __restrict__ lut,
                      const _Float16* __restrict__ bias,
                      _Float16*       __restrict__ out)
{
    __shared__ __align__(16) unsigned char smem[BM * LDA * 2 + BN * LDB * 2 + BN * 16 * 2];
    _Float16* As   = (_Float16*)smem;
    _Float16* Bs   = (_Float16*)(smem + BM * LDA * 2);
    _Float16* lutS = (_Float16*)(smem + BM * LDA * 2 + BN * LDB * 2);
    _Float16* Cs   = (_Float16*)smem;

    const int tid  = threadIdx.x;
    const int lane = tid & 31;
    const int wave = tid >> 5;
    const int r    = lane & 15;
    const int hi   = lane >> 4;
    const int wm   = wave & 3;
    const int wn   = wave >> 2;

    const int m0 = blockIdx.y * BM;
    const int n0 = blockIdx.x * BN;

    {
        const int row = tid >> 1, part = tid & 1;
        *(uint4*)(lutS + row * 16 + part * 8) =
            *(const uint4*)(lut + (size_t)(n0 + row) * 16 + part * 8);
    }

    const int rr  = tid >> 1;
    const int sub = tid & 1;
    const unsigned short* lrow = (const unsigned short*)lutS + rr * 16;

    v8f acc[4][4];
#pragma unroll
    for (int mi = 0; mi < 4; ++mi)
#pragma unroll
        for (int ni = 0; ni < 4; ++ni)
            acc[mi][ni] = (v8f){0.f, 0.f, 0.f, 0.f, 0.f, 0.f, 0.f, 0.f};

#pragma unroll 1
    for (int kt = 0; kt < IN_F; kt += BK) {
        __syncthreads();

#pragma unroll
        for (int it = 0; it < 8; ++it) {
            int lin = it * 256 + tid;
            int row = lin >> 3, ch = lin & 7;
            *(uint4*)(As + row * LDA + ch * 8) =
                *(const uint4*)(x + (size_t)(m0 + row) * IN_F + kt + ch * 8);
        }
        {
            const int* q = qw + (size_t)(n0 + rr) * WORDS + (kt >> 5) + sub;
            unsigned int cz[4];
            codes32((unsigned int)q[0], (unsigned int)q[PLANE],
                    (unsigned int)q[2 * PLANE], (unsigned int)q[3 * PLANE], cz);
            _Float16* brow = Bs + rr * LDB + sub * 32;
#pragma unroll
            for (int d = 0; d < 4; ++d) {
#pragma unroll
                for (int n = 0; n < 8; n += 2)
                    *(unsigned int*)(brow + 8 * d + n) = lut_pair(lrow, cz[d], n);
            }
        }

        const int ktn = (kt + BK < IN_F) ? (kt + BK) : kt;
        __builtin_prefetch(x  + (size_t)(m0 + 2 * rr + sub) * IN_F + ktn, 0, 0);
        __builtin_prefetch(qw + (size_t)(n0 + rr) * WORDS + (ktn >> 5) + sub, 0, 0);

        __syncthreads();

#pragma unroll
        for (int ks = 0; ks < BK; ks += 32) {
            Frag a[4], b[4];
#pragma unroll
            for (int mi = 0; mi < 4; ++mi) {
                const _Float16* ap = As + (wm * 64 + mi * 16 + r) * LDA + ks + hi * 8;
                a[mi].q[0] = *(const uint4*)ap;
                a[mi].q[1] = *(const uint4*)(ap + 16);
            }
#pragma unroll
            for (int ni = 0; ni < 4; ++ni) {
                const _Float16* bp = Bs + (wn * 64 + ni * 16 + r) * LDB + ks + hi * 8;
                b[ni].q[0] = *(const uint4*)bp;
                b[ni].q[1] = *(const uint4*)(bp + 16);
            }
#pragma unroll
            for (int mi = 0; mi < 4; ++mi)
#pragma unroll
                for (int ni = 0; ni < 4; ++ni)
                    acc[mi][ni] = __builtin_amdgcn_wmma_f32_16x16x32_f16(
                        false, a[mi].v, false, b[ni].v,
                        (short)0, acc[mi][ni], false, false);
        }
    }

    float bb[4];
#pragma unroll
    for (int ni = 0; ni < 4; ++ni)
        bb[ni] = (float)bias[n0 + wn * 64 + ni * 16 + r];

#pragma unroll 1
    for (int ph = 0; ph < 2; ++ph) {
        __syncthreads();
        if ((wm >> 1) == ph) {
#pragma unroll
            for (int mi = 0; mi < 4; ++mi)
#pragma unroll
                for (int ni = 0; ni < 4; ++ni)
#pragma unroll
                    for (int i = 0; i < 8; ++i) {
                        const int rowl = (wm & 1) * 64 + mi * 16 + hi * 8 + i;
                        const int col  = wn * 64 + ni * 16 + r;
                        Cs[rowl * LDC + col] = (_Float16)(acc[mi][ni][i] + bb[ni]);
                    }
        }
        __syncthreads();
#pragma unroll
        for (int it = 0; it < 8; ++it) {
            int lin = it * 256 + tid;
            int row = lin >> 4, ch = lin & 15;
            *(uint4*)(out + (size_t)(m0 + ph * 128 + row) * OUT_F + n0 + ch * 8) =
                *(const uint4*)(Cs + row * LDC + ch * 8);
        }
    }
}

extern "C" void kernel_launch(void* const* d_in, const int* in_sizes, int n_in,
                              void* d_out, int out_size, void* d_ws, size_t ws_size,
                              hipStream_t stream) {
    const _Float16* x    = (const _Float16*)d_in[0];
    const int*      qw   = (const int*)d_in[1];
    const _Float16* lut  = (const _Float16*)d_in[2];
    const _Float16* bias = (const _Float16*)d_in[3];
    _Float16* out = (_Float16*)d_out;

    const size_t w_bytes = (size_t)OUT_F * IN_F * sizeof(_Float16);  // ~90 MB
    dim3 grid(OUT_F / BN, (2 * 2048) / BM);   // (86, 16)

    if (ws_size >= w_bytes) {
        _Float16* wsW = (_Float16*)d_ws;
        apl_dequant_kernel<<<OUT_F / 2, 256, 0, stream>>>(qw, lut, wsW);
        apl_gemm_kernel<<<grid, 256, 0, stream>>>(x, wsW, bias, out);
    } else {
        apl_fused_kernel<<<grid, 256, 0, stream>>>(x, qw, lut, bias, out);
    }
}
